// LayerNormLSTM_30416958390862
// MI455X (gfx1250) — compile-verified, matrix operand order checked by reading, not run on hardware
//
#include <hip/hip_runtime.h>

typedef __attribute__((ext_vector_type(16))) __bf16 v16bf;
typedef __attribute__((ext_vector_type(8)))  __bf16 v8bf;
typedef __attribute__((ext_vector_type(8)))  float  v8f;

#define T_LEN   512
#define BATCH   64
#define IN_SIZE 1024
#define HID     1024
#define GATES   (4 * HID)          // 4096
#define MTOT    (T_LEN * BATCH)    // 32768
#define EPSV    1e-6f
// Padded row stride for the 16 x 4096 f32 accumulator tile in LDS.
// Pad = 20 floats: stats banks {sub, sub+20} disjoint; epilogue halves differ
// by 8*20 mod 64 = 32 banks -> conflict-free.
#define ACC_STRIDE (GATES + 20)    // 4116

__device__ __forceinline__ float sigmoidf(float x) {
    return 1.0f / (1.0f + __expf(-x));
}

// Build a v16bf WMMA fragment from two 16-byte chunks.
// Per ISA 7.12.2 (16-bit A 16x32 / B 32x16): elements 0..7 hold K=kbase..kbase+7,
// elements 8..15 hold K=kbase+16..kbase+23 (kbase = 0 or 8 by lane half).
__device__ __forceinline__ v16bf load_frag(const __bf16* lo_p, const __bf16* hi_p) {
    v8bf lo = *(const v8bf*)lo_p;
    v8bf hi = *(const v8bf*)hi_p;
    v16bf r;
#pragma unroll
    for (int i = 0; i < 8; ++i) { r[i] = lo[i]; r[i + 8] = hi[i]; }
    return r;
}

// ---------------- Kernel 1: convert Wi (4096x1024 f32) -> bf16 ----------------
__global__ void wi_to_bf16_kernel(const float* __restrict__ Wi,
                                  __bf16* __restrict__ WiB) {
    size_t i4 = (size_t)blockIdx.x * blockDim.x + threadIdx.x; // float4 index
    float4 v = ((const float4*)Wi)[i4];
    __bf16* d = WiB + i4 * 4;
    d[0] = (__bf16)v.x; d[1] = (__bf16)v.y; d[2] = (__bf16)v.z; d[3] = (__bf16)v.w;
}

// ---------------- Kernel 2: h2h = LN(bh)*g_h2h + b_h2h  (length 4096) --------
__global__ void h2h_prep_kernel(const float* __restrict__ bh,
                                const float* __restrict__ g_h2h,
                                const float* __restrict__ b_h2h,
                                float* __restrict__ h2h) {
    __shared__ float rs[256], rq[256];
    __shared__ float s_mu, s_rstd;
    int t = threadIdx.x;
    float s = 0.f, q = 0.f;
    for (int i = 0; i < GATES / 256; ++i) {
        float v = bh[t + i * 256];
        s += v; q += v * v;
    }
    rs[t] = s; rq[t] = q;
    __syncthreads();
    for (int off = 128; off > 0; off >>= 1) {
        if (t < off) { rs[t] += rs[t + off]; rq[t] += rq[t + off]; }
        __syncthreads();
    }
    if (t == 0) {
        float mu  = rs[0] / (float)GATES;
        float var = (rq[0] - (float)GATES * mu * mu) / (float)(GATES - 1);
        s_mu = mu; s_rstd = rsqrtf(var + EPSV);
    }
    __syncthreads();
    float mu = s_mu, rstd = s_rstd;
    for (int i = 0; i < GATES / 256; ++i) {
        int idx = t + i * 256;
        h2h[idx] = g_h2h[idx] * ((bh[idx] - mu) * rstd) + b_h2h[idx];
    }
}

// ---------------- Kernel 3: fused GEMM + LN + gates + LN + output ------------
// One workgroup = 16 rows (one WMMA M-tile) x full N=4096.
// 256 threads = 8 waves; wave w owns N-tiles [w*32, w*32+32), processed 4 at a
// time (4 independent WMMA accumulator chains; one A fragment feeds 4 WMMAs).
__global__ void __launch_bounds__(256)
lnlstm_fused_kernel(const float* __restrict__ x,        // (MTOT, IN_SIZE)
                    const __bf16* __restrict__ WiB,     // (GATES, IN_SIZE) bf16
                    const float* __restrict__ bi,       // (GATES)
                    const float* __restrict__ h2h,      // (GATES)
                    const float* __restrict__ g_i2h,
                    const float* __restrict__ b_i2h,
                    const float* __restrict__ g_cell,
                    const float* __restrict__ b_cell,
                    float* __restrict__ out) {          // MTOT*HID then BATCH*HID
    __shared__ float accLds[16 * ACC_STRIDE];                   // ~257 KB
    __shared__ __align__(16) __bf16 aLds[16 * IN_SIZE];         // 32 KB
    __shared__ float redS[16][16], redQ[16][16];
    __shared__ float muA[16], rsA[16], muC[16], rsC[16];

    const int t    = threadIdx.x;
    const int wave = t >> 5;
    const int lane = t & 31;
    const int m0   = blockIdx.x * 16;

    // ---- Stage A tile: x[m0:m0+16, :] f32 -> bf16 in LDS ----
    {
        const float* xg = x + (size_t)m0 * IN_SIZE;
        for (int i = 0; i < 16; ++i) {
            int i4 = t + i * 256;                  // 4096 float4 = 16x1024 floats
            float4 v = ((const float4*)xg)[i4];
            __bf16* d = &aLds[i4 * 4];
            d[0] = (__bf16)v.x; d[1] = (__bf16)v.y;
            d[2] = (__bf16)v.z; d[3] = (__bf16)v.w;
        }
    }
    __syncthreads();

    // ---- GEMM: 32 N-tiles per wave, 4 concurrent accumulator chains ----
    const int nrow  = lane & 15;        // M index (A) / N index (B, C)
    const int khalf = (lane >> 4) * 8;  // K sub-base per lane half
    const int mhalf = (lane >> 4) * 8;  // M offset of C rows for this lane half
    const __bf16* arow = &aLds[(size_t)nrow * IN_SIZE];

    for (int grp = 0; grp < 8; ++grp) {
        const int nt0 = wave * 32 + grp * 4;
        const __bf16* brow0 = WiB + ((size_t)(nt0 * 16 + nrow)) * IN_SIZE;
        const __bf16* brow1 = brow0 + (size_t)16 * IN_SIZE;
        const __bf16* brow2 = brow1 + (size_t)16 * IN_SIZE;
        const __bf16* brow3 = brow2 + (size_t)16 * IN_SIZE;
        v8f acc0 = {}, acc1 = {}, acc2 = {}, acc3 = {};
#pragma unroll 2
        for (int k0 = 0; k0 < IN_SIZE; k0 += 32) {
            v16bf af = load_frag(arow + k0 + khalf, arow + k0 + khalf + 16);
            v16bf b0 = load_frag(brow0 + k0 + khalf, brow0 + k0 + khalf + 16);
            v16bf b1 = load_frag(brow1 + k0 + khalf, brow1 + k0 + khalf + 16);
            v16bf b2 = load_frag(brow2 + k0 + khalf, brow2 + k0 + khalf + 16);
            v16bf b3 = load_frag(brow3 + k0 + khalf, brow3 + k0 + khalf + 16);
            acc0 = __builtin_amdgcn_wmma_f32_16x16x32_bf16(
                false, af, false, b0, (short)0, acc0, false, false);
            acc1 = __builtin_amdgcn_wmma_f32_16x16x32_bf16(
                false, af, false, b1, (short)0, acc1, false, false);
            acc2 = __builtin_amdgcn_wmma_f32_16x16x32_bf16(
                false, af, false, b2, (short)0, acc2, false, false);
            acc3 = __builtin_amdgcn_wmma_f32_16x16x32_bf16(
                false, af, false, b3, (short)0, acc3, false, false);
        }
        // Epilogue: add bi, spill to LDS acc tile (C layout: m = r + mhalf, n)
#pragma unroll
        for (int q = 0; q < 4; ++q) {
            const int n = (nt0 + q) * 16 + nrow;
            const float biv = bi[n];
            const v8f acc = (q == 0) ? acc0 : (q == 1) ? acc1 : (q == 2) ? acc2 : acc3;
#pragma unroll
            for (int r = 0; r < 8; ++r)
                accLds[(r + mhalf) * ACC_STRIDE + n] = acc[r] + biv;
        }
    }
    __syncthreads();

    // ---- LN stats over full 4096 gate dim (pre-norm preact = Y + bi) ----
    {
        int row = t >> 4, sub = t & 15;
        const float* p = &accLds[row * ACC_STRIDE];
        float s = 0.f, q = 0.f;
        for (int i = sub; i < GATES; i += 16) { float v = p[i]; s += v; q += v * v; }
        redS[row][sub] = s; redQ[row][sub] = q;
    }
    __syncthreads();
    if (t < 16) {
        float s = 0.f, q = 0.f;
        for (int i = 0; i < 16; ++i) { s += redS[t][i]; q += redQ[t][i]; }
        float mu  = s / (float)GATES;
        float var = (q - (float)GATES * mu * mu) / (float)(GATES - 1);
        muA[t] = mu; rsA[t] = rsqrtf(var + EPSV);
    }
    __syncthreads();

    // ---- Gates: c = (1-f)*g, stored into the dead input-gate columns ----
    for (int it = 0; it < 64; ++it) {
        int idx = t + it * 256;            // 16 * 1024 items
        int m = idx >> 10, j = idx & 1023;
        float mu = muA[m], rs = rsA[m];
        int cf = HID + j, cg = 3 * HID + j;
        float pf = g_i2h[cf] * ((accLds[m * ACC_STRIDE + cf] - mu) * rs)
                 + b_i2h[cf] + h2h[cf];
        float pg = g_i2h[cg] * ((accLds[m * ACC_STRIDE + cg] - mu) * rs)
                 + b_i2h[cg] + h2h[cg];
        float f = sigmoidf(pf);
        float g = tanhf(pg);
        accLds[m * ACC_STRIDE + j] = (1.0f - f) * g;
    }
    __syncthreads();

    // ---- LN stats for c over H=1024 ----
    {
        int row = t >> 4, sub = t & 15;
        const float* p = &accLds[row * ACC_STRIDE];
        float s = 0.f, q = 0.f;
        for (int i = sub; i < HID; i += 16) { float v = p[i]; s += v; q += v * v; }
        redS[row][sub] = s; redQ[row][sub] = q;
    }
    __syncthreads();
    if (t < 16) {
        float s = 0.f, q = 0.f;
        for (int i = 0; i < 16; ++i) { s += redS[t][i]; q += redQ[t][i]; }
        float mu  = s / (float)HID;
        float var = (q - (float)HID * mu * mu) / (float)(HID - 1);
        muC[t] = mu; rsC[t] = rsqrtf(var + EPSV);
    }
    __syncthreads();

    // ---- h = o * tanh(LN(c)); write h_t and (for t == T-1) h_last ----
    const size_t TBH = (size_t)MTOT * HID;
    for (int it = 0; it < 64; ++it) {
        int idx = t + it * 256;
        int m = idx >> 10, j = idx & 1023;
        float cn = g_cell[j] * ((accLds[m * ACC_STRIDE + j] - muC[m]) * rsC[m])
                 + b_cell[j];
        int co = 2 * HID + j;
        float po = g_i2h[co] * ((accLds[m * ACC_STRIDE + co] - muA[m]) * rsA[m])
                 + b_i2h[co] + h2h[co];
        float h = sigmoidf(po) * tanhf(cn);
        int gm = m0 + m;
        out[(size_t)gm * HID + j] = h;
        if (gm >= (T_LEN - 1) * BATCH)
            out[TBH + (size_t)(gm - (T_LEN - 1) * BATCH) * HID + j] = h;
    }
}

extern "C" void kernel_launch(void* const* d_in, const int* in_sizes, int n_in,
                              void* d_out, int out_size, void* d_ws, size_t ws_size,
                              hipStream_t stream) {
    const float* x      = (const float*)d_in[0];
    const float* Wi     = (const float*)d_in[1];
    const float* bi     = (const float*)d_in[2];
    // d_in[3] = Wh (dead: h0 == 0)
    const float* bh     = (const float*)d_in[4];
    const float* g_i2h  = (const float*)d_in[5];
    const float* b_i2h  = (const float*)d_in[6];
    const float* g_h2h  = (const float*)d_in[7];
    const float* b_h2h  = (const float*)d_in[8];
    const float* g_cell = (const float*)d_in[9];
    const float* b_cell = (const float*)d_in[10];
    float* out = (float*)d_out;

    // Workspace layout: [0, 8MB) Wi in bf16; [8MB, 8MB+16KB) h2h vector (f32)
    __bf16* WiB = (__bf16*)d_ws;
    float*  h2h = (float*)((char*)d_ws + (size_t)GATES * IN_SIZE * sizeof(__bf16));

    // Wi f32 -> bf16 (4,194,304 elems; 4 per thread via float4)
    wi_to_bf16_kernel<<<(GATES * IN_SIZE) / (256 * 4), 256, 0, stream>>>(Wi, WiB);

    // h2h vector (single block)
    h2h_prep_kernel<<<1, 256, 0, stream>>>(bh, g_h2h, b_h2h, h2h);

    // Fused GEMM + double LayerNorm + gates
    lnlstm_fused_kernel<<<MTOT / 16, 256, 0, stream>>>(
        x, WiB, bi, h2h, g_i2h, b_i2h, g_cell, b_cell, out);
}
